// SelfAttention_33200097198570
// MI455X (gfx1250) — matrix operand order repaired and checked by reading, not compiled, not run on hardware
//
#include <hip/hip_runtime.h>
#include <hip/hip_bf16.h>

// Problem constants (B, T, E, H, S)
#define B_ 4
#define T_ 2048
#define E_ 1024
#define H_ 16
#define S_ 64
static constexpr float QK_SCALE = 0.17677669529663687f; // E^-0.25 (applied to q AND k)

typedef __bf16 bf16_t;
typedef bf16_t bf16x16 __attribute__((ext_vector_type(16)));
typedef float  f32x8   __attribute__((ext_vector_type(8)));

union Frag {                 // one WMMA 16-bf16 operand fragment (32B / lane)
  bf16x16 v;
  uint4   q[2];
  unsigned short h[16];
};
union U4 {
  uint4 q;
  unsigned short h[8];
};

__device__ __forceinline__ unsigned short f2bf(float f) {
  unsigned int u = __float_as_uint(f);
  u += 0x7FFFu + ((u >> 16) & 1u);   // round-to-nearest-even
  return (unsigned short)(u >> 16);
}

// Low 32 bits of a generic pointer into LDS == LDS byte offset (aperture in
// the high bits per CDNA5 aperture rules).
__device__ __forceinline__ unsigned lds_off32(const void* p) {
  return (unsigned)(uintptr_t)p;
}

// Per-lane async copy: 16B global -> 16B LDS, tracked by ASYNCcnt, no VGPR
// staging (GLOBAL_LOAD_ASYNC_TO_LDS_B128, CDNA5 ISA).
__device__ __forceinline__ void async_copy_b128(unsigned lds_off, const void* g) {
  asm volatile("global_load_async_to_lds_b128 %0, %1, off"
               :
               : "v"(lds_off), "v"((unsigned long long)(uintptr_t)g)
               : "memory");
}

__device__ __forceinline__ void wait_async0() {   // all async ops landed
#if __has_builtin(__builtin_amdgcn_s_wait_asynccnt)
  __builtin_amdgcn_s_wait_asynccnt(0);
#else
  asm volatile("s_wait_asynccnt 0x0" ::: "memory");
#endif
}
__device__ __forceinline__ void wait_async4() {   // oldest (cnt-4) ops landed
#if __has_builtin(__builtin_amdgcn_s_wait_asynccnt)
  __builtin_amdgcn_s_wait_asynccnt(4);
#else
  asm volatile("s_wait_asynccnt 0x4" ::: "memory");
#endif
}

// ---------------------------------------------------------------------------
// Kernel 1: fp32 -> bf16 elementwise (x), 8 elements / thread, vectorized
// ---------------------------------------------------------------------------
__global__ void cvt_f32_bf16(const float* __restrict__ src,
                             unsigned short* __restrict__ dst, int n8) {
  int i = blockIdx.x * blockDim.x + threadIdx.x;
  if (i >= n8) return;
  const float4* s4 = (const float4*)src;
  float4 a = s4[2 * i + 0];
  float4 b = s4[2 * i + 1];
  U4 o;
  o.h[0] = f2bf(a.x); o.h[1] = f2bf(a.y); o.h[2] = f2bf(a.z); o.h[3] = f2bf(a.w);
  o.h[4] = f2bf(b.x); o.h[5] = f2bf(b.y); o.h[6] = f2bf(b.z); o.h[7] = f2bf(b.w);
  ((uint4*)dst)[i] = o.q;
}

// ---------------------------------------------------------------------------
// Kernel 2: W (K x N fp32, row-major) -> Wt (N x K bf16, row-major)
// N x K layout makes WMMA B-fragments contiguous per-lane loads.
// ---------------------------------------------------------------------------
__global__ void transpose_w_bf16(const float* __restrict__ W,
                                 unsigned short* __restrict__ Wt) {
  __shared__ float tile[32][33];
  const int tx = threadIdx.x, ty = threadIdx.y;           // 32 x 8
  const int nb = blockIdx.x * 32;                         // N base
  const int kb = blockIdx.y * 32;                         // K base
  for (int i = 0; i < 32; i += 8)
    tile[ty + i][tx] = W[(size_t)(kb + ty + i) * E_ + nb + tx];
  __syncthreads();
  for (int i = 0; i < 32; i += 8)
    Wt[(size_t)(nb + ty + i) * E_ + kb + tx] = f2bf(tile[tx][ty + i]);
}

// ---------------------------------------------------------------------------
// Kernel 3: bf16 WMMA GEMM  C[M,N] = A[M,K] * Bt[N,K]^T
//   block: 256 thr (8 waves), tile 128x128, K-step 32, wave tile 32x64.
//   Double-buffered GLOBAL_LOAD_ASYNC_TO_LDS_B128 pipeline: tile i+1 streams
//   into LDS (ASYNCcnt) while tile i feeds the WMMAs; s_wait_asynccnt 4
//   releases only the oldest tile (async ops complete in order).
//   mode 0: C fp32 row-major + bias          (output projection)
//   mode 1: C bf16 * scale -> (B,H,T,S)      (q/k projection, pre-scaled)
// ---------------------------------------------------------------------------
__global__ __launch_bounds__(256) void gemm_bf16_wmma(
    const unsigned short* __restrict__ A,   // M x 1024 bf16
    const unsigned short* __restrict__ Bt,  // 1024 x 1024 bf16 (N x K)
    float* __restrict__ Cf,                 // mode 0
    const float* __restrict__ bias,         // mode 0
    unsigned short* __restrict__ Cb,        // mode 1
    float scale, int mode) {
  constexpr int K_   = E_;
  constexpr int LDA  = 40;                  // padded LDS stride (elements)
  constexpr int TILE = 128 * LDA;           // elements per buffer
  constexpr int NT   = K_ / 32;             // 32 K-tiles
  __shared__ __align__(16) unsigned short Ash[2 * TILE];
  __shared__ __align__(16) unsigned short Bsh[2 * TILE];

  const int tid  = threadIdx.x;
  const int lane = tid & 31;                // wave32
  const int wave = tid >> 5;                // 0..7
  const int wm   = wave & 3;                // 4 waves along M
  const int wn   = wave >> 2;               // 2 waves along N
  const int l16  = lane & 15;
  const int lh   = lane >> 4;

  const int mBase = blockIdx.x * 128;
  const int nBase = blockIdx.y * 128;

  f32x8 acc[2][4];
  for (int mi = 0; mi < 2; ++mi)
    for (int ni = 0; ni < 4; ++ni)
      acc[mi][ni] = f32x8{0.f, 0.f, 0.f, 0.f, 0.f, 0.f, 0.f, 0.f};

  const int ldRow = tid >> 1;               // 0..127
  const int ldCol = (tid & 1) * 16;         // 0 / 16

  const unsigned aDst0 = lds_off32(&Ash[ldRow * LDA + ldCol]);
  const unsigned aDst1 = lds_off32(&Ash[ldRow * LDA + ldCol + 8]);
  const unsigned bDst0 = lds_off32(&Bsh[ldRow * LDA + ldCol]);
  const unsigned bDst1 = lds_off32(&Bsh[ldRow * LDA + ldCol + 8]);
  const unsigned short* gA = A  + (size_t)(mBase + ldRow) * K_ + ldCol;
  const unsigned short* gB = Bt + (size_t)(nBase + ldRow) * K_ + ldCol;

  auto issue_tile = [&](int k0, int buf) {
    const unsigned boff = (unsigned)buf * (TILE * 2u);   // bytes
    async_copy_b128(aDst0 + boff, gA + k0);
    async_copy_b128(aDst1 + boff, gA + k0 + 8);
    async_copy_b128(bDst0 + boff, gB + k0);
    async_copy_b128(bDst1 + boff, gB + k0 + 8);
  };

  issue_tile(0, 0);                         // prologue: tile 0 in flight
  for (int it = 0; it < NT; ++it) {
    const int cur = it & 1;
    if (it + 1 < NT) {
      issue_tile((it + 1) * 32, cur ^ 1);   // overlap next tile with compute
      if (it + 2 < NT) {                    // warm L2 two tiles ahead
        __builtin_prefetch(gA + (it + 2) * 32, 0, 0);
        __builtin_prefetch(gB + (it + 2) * 32, 0, 0);
      }
      wait_async4();                        // oldest 4 ops (tile it) landed
    } else {
      wait_async0();
    }
    __syncthreads();                        // tile it visible to all waves

    const unsigned short* As = &Ash[cur * TILE];
    const unsigned short* Bs = &Bsh[cur * TILE];
    Frag a[2], b[4];
    for (int mi = 0; mi < 2; ++mi) {        // A-frag: row = lane%16, K chunks
      const int row = wm * 32 + mi * 16 + l16;
      const int klo = lh * 8;
      a[mi].q[0] = *(const uint4*)&As[row * LDA + klo];
      a[mi].q[1] = *(const uint4*)&As[row * LDA + 16 + klo];
    }
    for (int ni = 0; ni < 4; ++ni) {        // B-frag: col = lane%16, K chunks
      const int nrow = wn * 64 + ni * 16 + l16;
      const int kc   = lh * 8;
      b[ni].q[0] = *(const uint4*)&Bs[nrow * LDA + kc];
      b[ni].q[1] = *(const uint4*)&Bs[nrow * LDA + kc + 16];
    }
    for (int mi = 0; mi < 2; ++mi)
      for (int ni = 0; ni < 4; ++ni)
        acc[mi][ni] = __builtin_amdgcn_wmma_f32_16x16x32_bf16(
            false, a[mi].v, false, b[ni].v, (short)0, acc[mi][ni], false, false);
    __syncthreads();                        // buffer free for tile it+2
  }

  // Epilogue: C-frag row = r + 8*(lane/16), col = lane%16
  for (int mi = 0; mi < 2; ++mi) {
    for (int ni = 0; ni < 4; ++ni) {
      const int col = nBase + wn * 64 + ni * 16 + l16;
      for (int r = 0; r < 8; ++r) {
        const int row = mBase + wm * 32 + mi * 16 + r + lh * 8;
        const float v = acc[mi][ni][r];
        if (mode == 0) {
          Cf[(size_t)row * E_ + col] = v + bias[col];
        } else {
          const int bi = row >> 11, t = row & (T_ - 1);   // row = b*T + t
          const int h  = col >> 6,  s = col & (S_ - 1);   // col = h*S + s
          Cb[((size_t)(bi * H_ + h) * T_ + t) * S_ + s] = f2bf(v * scale);
        }
      }
    }
  }
}

// ---------------------------------------------------------------------------
// Kernel 4: flash attention, one block = 64 queries of one (b,h)
//   4 waves x 16 queries; double-buffered 64-key tiles (K doubles as V).
//   Ksh  : 64 keys x 64 d (row-major), filled via async-to-LDS (ASYNCcnt)
//   KTsh : 64 d x 64 keys (transposed on the fly from a mirrored normal load)
//   Psh  : wave-private 16 x 64 bf16 P -> A-frags for the second WMMA
// ---------------------------------------------------------------------------
__global__ __launch_bounds__(128) void flash_attn_wmma(
    const unsigned short* __restrict__ Q,   // (B*H, T, S) bf16, pre-scaled
    const unsigned short* __restrict__ K,   // (B*H, T, S) bf16, pre-scaled (= V)
    unsigned short* __restrict__ O) {       // (B, T, E) bf16
  constexpr int LDK   = 72;                 // padded stride
  constexpr int KTILE = 64 * LDK;           // elements per buffer
  constexpr int NKT   = T_ / 64;            // 32 key tiles
  __shared__ __align__(16) unsigned short Ksh [2 * KTILE];
  __shared__ __align__(16) unsigned short KTsh[2 * KTILE];
  __shared__ __align__(16) unsigned short Psh [4 * 16 * LDK];

  const int tid  = threadIdx.x;
  const int lane = tid & 31;
  const int wave = tid >> 5;                // 0..3
  const int l16  = lane & 15;
  const int lh   = lane >> 4;

  const int bh   = blockIdx.x;              // 0..B*H-1
  const int qblk = blockIdx.y;              // 0..T/64-1
  const size_t baseQK = (size_t)bh * T_ * S_;

  // Q fragments for this wave's 16 queries (kept resident: 2 frags = 16 VGPRs)
  Frag qf[2];
  {
    const int qrow = qblk * 64 + wave * 16 + l16;
    const unsigned short* qp = Q + baseQK + (size_t)qrow * S_;
    const int klo = lh * 8;
    qf[0].q[0] = *(const uint4*)(qp + klo);
    qf[0].q[1] = *(const uint4*)(qp + 16 + klo);
    qf[1].q[0] = *(const uint4*)(qp + 32 + klo);
    qf[1].q[1] = *(const uint4*)(qp + 48 + klo);
  }

  f32x8 o[4];
  for (int dn = 0; dn < 4; ++dn)
    o[dn] = f32x8{0.f, 0.f, 0.f, 0.f, 0.f, 0.f, 0.f, 0.f};
  float mrow[8], lrow[8];
  for (int r = 0; r < 8; ++r) { mrow[r] = -1e30f; lrow[r] = 0.f; }

  // Row-major tile -> async straight to LDS (4 ops/wave); the same lines are
  // pulled through VGPRs once to build the transposed copy for the PV WMMA.
  auto load_tile = [&](int kb, int buf) {
    unsigned short* kt = &KTsh[buf * KTILE];
    for (int j = 0; j < 4; ++j) {
      const int u   = tid * 4 + j;
      const int key = u >> 3;
      const int d0  = (u & 7) * 8;
      const unsigned short* gp = K + baseQK + (size_t)(kb + key) * S_ + d0;
      async_copy_b128(lds_off32(&Ksh[buf * KTILE + key * LDK + d0]), gp);
      U4 val;
      val.q = *(const uint4*)gp;
      for (int ii = 0; ii < 8; ++ii)
        kt[(d0 + ii) * LDK + key] = val.h[ii];
    }
  };

  load_tile(0, 0);                          // prologue: tile 0 in flight
  for (int it = 0; it < NKT; ++it) {
    const int cur = it & 1;
    if (it + 1 < NKT) {
      load_tile((it + 1) * 64, cur ^ 1);    // overlap next key tile
      wait_async4();                        // tile it's 4 async ops landed
    } else {
      wait_async0();
    }
    __syncthreads();

    const unsigned short* Ks  = &Ksh [cur * KTILE];
    const unsigned short* KTs = &KTsh[cur * KTILE];

    // S = Q K^T : 16 queries x 64 keys per wave (8 WMMA)
    f32x8 sf[4];
    for (int ks = 0; ks < 4; ++ks) {
      Frag b0, b1;
      const int key  = ks * 16 + l16;
      const int doff = lh * 8;
      b0.q[0] = *(const uint4*)&Ks[key * LDK + doff];
      b0.q[1] = *(const uint4*)&Ks[key * LDK + doff + 16];
      b1.q[0] = *(const uint4*)&Ks[key * LDK + 32 + doff];
      b1.q[1] = *(const uint4*)&Ks[key * LDK + 32 + doff + 16];
      f32x8 z = f32x8{0.f, 0.f, 0.f, 0.f, 0.f, 0.f, 0.f, 0.f};
      z = __builtin_amdgcn_wmma_f32_16x16x32_bf16(false, qf[0].v, false, b0.v,
                                                  (short)0, z, false, false);
      z = __builtin_amdgcn_wmma_f32_16x16x32_bf16(false, qf[1].v, false, b1.v,
                                                  (short)0, z, false, false);
      sf[ks] = z;
    }

    // Online softmax: row reductions across the 16-lane half via xor shuffles
    for (int r = 0; r < 8; ++r) {
      float mx = fmaxf(fmaxf(sf[0][r], sf[1][r]), fmaxf(sf[2][r], sf[3][r]));
      for (int off = 1; off < 16; off <<= 1)
        mx = fmaxf(mx, __shfl_xor(mx, off, 32));
      const float mn    = fmaxf(mrow[r], mx);
      const float alpha = __expf(mrow[r] - mn);
      float psum = 0.f;
      for (int f = 0; f < 4; ++f) {
        const float p = __expf(sf[f][r] - mn);
        sf[f][r] = p;
        psum += p;
      }
      for (int off = 1; off < 16; off <<= 1)
        psum += __shfl_xor(psum, off, 32);
      lrow[r] = lrow[r] * alpha + psum;
      mrow[r] = mn;
      for (int dn = 0; dn < 4; ++dn) o[dn][r] *= alpha;
    }

    // P (bf16) -> wave-private LDS, then back as A-fragments
    unsigned short* pw = &Psh[wave * 16 * LDK];
    for (int f = 0; f < 4; ++f)
      for (int r = 0; r < 8; ++r)
        pw[(r + lh * 8) * LDK + f * 16 + l16] = f2bf(sf[f][r]);

    Frag pa[2];
    {
      const int klo = lh * 8;
      pa[0].q[0] = *(const uint4*)&pw[l16 * LDK + klo];
      pa[0].q[1] = *(const uint4*)&pw[l16 * LDK + 16 + klo];
      pa[1].q[0] = *(const uint4*)&pw[l16 * LDK + 32 + klo];
      pa[1].q[1] = *(const uint4*)&pw[l16 * LDK + 48 + klo];
    }
    // O += P V  (V = key tile, read transposed; 8 WMMA)
    for (int dn = 0; dn < 4; ++dn) {
      Frag v0, v1;
      const int d   = dn * 16 + l16;
      const int kc0 = lh * 8;
      v0.q[0] = *(const uint4*)&KTs[d * LDK + kc0];
      v0.q[1] = *(const uint4*)&KTs[d * LDK + kc0 + 16];
      v1.q[0] = *(const uint4*)&KTs[d * LDK + 32 + kc0];
      v1.q[1] = *(const uint4*)&KTs[d * LDK + 32 + kc0 + 16];
      o[dn] = __builtin_amdgcn_wmma_f32_16x16x32_bf16(false, pa[0].v, false, v0.v,
                                                      (short)0, o[dn], false, false);
      o[dn] = __builtin_amdgcn_wmma_f32_16x16x32_bf16(false, pa[1].v, false, v1.v,
                                                      (short)0, o[dn], false, false);
    }
    __syncthreads();                        // buffer free for tile it+2
  }

  // Finalize: O /= l, store bf16 into (B, T, E) for the output projection
  const int b = bh >> 4, h = bh & (H_ - 1);
  for (int dn = 0; dn < 4; ++dn) {
    const int col = h * S_ + dn * 16 + l16;
    for (int r = 0; r < 8; ++r) {
      const int qg = qblk * 64 + wave * 16 + r + lh * 8;
      const float v = o[dn][r] / lrow[r];
      O[((size_t)(b * T_ + qg)) * E_ + col] = f2bf(v);
    }
  }
}

// ---------------------------------------------------------------------------
// Host launcher
// ---------------------------------------------------------------------------
extern "C" void kernel_launch(void* const* d_in, const int* in_sizes, int n_in,
                              void* d_out, int out_size, void* d_ws, size_t ws_size,
                              hipStream_t stream) {
  const float* x  = (const float*)d_in[0];
  const float* Wk = (const float*)d_in[1];
  const float* Wq = (const float*)d_in[2];
  const float* Wo = (const float*)d_in[3];
  const float* bo = (const float*)d_in[4];

  const size_t M  = (size_t)B_ * T_;        // 8192
  char* ws = (char*)d_ws;
  size_t off = 0;
  auto carve = [&](size_t bytes) {
    char* p = ws + off;
    off += (bytes + 255) & ~(size_t)255;
    return p;
  };
  unsigned short* xb  = (unsigned short*)carve(M * E_ * 2);          // x bf16
  unsigned short* Wqt = (unsigned short*)carve((size_t)E_ * E_ * 2); // Wq^T bf16
  unsigned short* Wkt = (unsigned short*)carve((size_t)E_ * E_ * 2);
  unsigned short* Wot = (unsigned short*)carve((size_t)E_ * E_ * 2);
  unsigned short* qb  = (unsigned short*)carve(M * E_ * 2);          // (B,H,T,S)
  unsigned short* kbf = (unsigned short*)carve(M * E_ * 2);          // (B,H,T,S)
  unsigned short* ob  = (unsigned short*)carve(M * E_ * 2);          // (B,T,E)
  (void)ws_size; (void)in_sizes; (void)n_in; (void)out_size;

  // 1) x -> bf16
  {
    const int n8 = (int)(M * E_ / 8);
    cvt_f32_bf16<<<(n8 + 255) / 256, 256, 0, stream>>>(x, xb, n8);
  }
  // 2) weights -> transposed bf16
  {
    dim3 tb(32, 8), tg(E_ / 32, E_ / 32);
    transpose_w_bf16<<<tg, tb, 0, stream>>>(Wq, Wqt);
    transpose_w_bf16<<<tg, tb, 0, stream>>>(Wk, Wkt);
    transpose_w_bf16<<<tg, tb, 0, stream>>>(Wo, Wot);
  }
  // 3) projections (pre-scaled by E^-0.25; v == k per the reference bug)
  {
    dim3 grid((unsigned)(M / 128), E_ / 128);
    gemm_bf16_wmma<<<grid, 256, 0, stream>>>(xb, Wqt, nullptr, nullptr, qb,
                                             QK_SCALE, 1);
    gemm_bf16_wmma<<<grid, 256, 0, stream>>>(xb, Wkt, nullptr, nullptr, kbf,
                                             QK_SCALE, 1);
  }
  // 4) flash attention
  {
    dim3 grid(B_ * H_, T_ / 64);
    flash_attn_wmma<<<grid, 128, 0, stream>>>(qb, kbf, ob);
  }
  // 5) output projection + bias -> fp32 d_out
  {
    dim3 grid((unsigned)(M / 128), E_ / 128);
    gemm_bf16_wmma<<<grid, 256, 0, stream>>>(ob, Wot, (float*)d_out, bo, nullptr,
                                             1.0f, 0);
  }
}
